// Seq2Seq_BiLSTM_46557445489131
// MI455X (gfx1250) — compile-verified, matrix operand order checked by reading, not run on hardware
//
#include <hip/hip_runtime.h>
#include <hip/hip_bf16.h>
#include <stdint.h>

typedef __attribute__((ext_vector_type(16))) __bf16        v16bf;
typedef __attribute__((ext_vector_type(8)))  float         v8f;
typedef __attribute__((ext_vector_type(8)))  unsigned int  v8u;
typedef __attribute__((ext_vector_type(4)))  unsigned int  v4u;

#define BN 256   // batch
#define TT 96    // time steps
#define DD 64    // input dim
#define HH 256   // encoder hidden per dir
#define LL 5     // layers
#define PP 14    // decoder steps
#define BT 16    // batch tile per block

__device__ __forceinline__ unsigned short f2bf(float f) {
  unsigned int u = __float_as_uint(f);
  u += 0x7FFFu + ((u >> 16) & 1u);   // round-to-nearest-even
  return (unsigned short)(u >> 16);
}
__device__ __forceinline__ float sigm(float x) { return 1.f / (1.f + __expf(-x)); }
__device__ __forceinline__ float tanh_(float x) {
  float e = __expf(2.f * x);
  return 1.f - 2.f / (e + 1.f);
}

union AFrag { v4u q[2]; v8u o; };

// ---------------------------------------------------------------------------
// Pack Wcat = [Wih ; Whh] (K x N, N = gate columns) into WMMA B-fragment
// order: tile (kt*NT + nt) is 512 bf16, element = [lane*16 + slot],
// slot s -> v=s/2,pos=s&1, k_local = 2v + (v>=4?8:0) + 8*(lane>=16) + pos,
// n = nt*16 + (lane&15).  Hot loop then loads 32B/lane (2x b128).
// ---------------------------------------------------------------------------
__global__ void pack_wcat_kernel(const float* __restrict__ Wih,
                                 const float* __restrict__ Whh,
                                 int Din, int H, int NT, int KT,
                                 unsigned short* __restrict__ dst) {
  int total = KT * NT * 512;
  int idx = blockIdx.x * blockDim.x + threadIdx.x;
  if (idx >= total) return;
  int within = idx & 511;
  int tile   = idx >> 9;
  int nt = tile % NT;
  int kt = tile / NT;
  int lane = within >> 4;
  int s    = within & 15;
  int v = s >> 1, pos = s & 1;
  int hi = lane >> 4;
  int k = kt * 32 + 2 * v + (v >= 4 ? 8 : 0) + 8 * hi + pos;
  int n = nt * 16 + (lane & 15);
  float val = (k < Din) ? Wih[(size_t)n * Din + k] : Whh[(size_t)n * H + (k - Din)];
  dst[idx] = f2bf(val);
}

__global__ void bias_add_kernel(const float* __restrict__ a,
                                const float* __restrict__ b,
                                float* __restrict__ d, int n) {
  int i = blockIdx.x * blockDim.x + threadIdx.x;
  if (i < n) d[i] = a[i] + b[i];
}

// x (B,T,D) f32 -> X0 (T,B,D) bf16 ; also Z0 = bf16(x[:, T-1, :])
__global__ void x_prep_kernel(const float* __restrict__ x,
                              unsigned short* __restrict__ X0,
                              unsigned short* __restrict__ Z0) {
  int i = blockIdx.x * blockDim.x + threadIdx.x;
  if (i >= BN * TT * DD) return;
  int k = i % DD;
  int t = (i / DD) % TT;
  int b = i / (DD * TT);
  unsigned short v = f2bf(x[i]);
  X0[((size_t)t * BN + b) * DD + k] = v;
  if (t == TT - 1) Z0[b * DD + k] = v;
}

// ---------------------------------------------------------------------------
// Persistent encoder scan: one block = 16 batch rows x 1 direction, loops all
// 96 timesteps.  gates = [x_t ; h] @ Wcat + bias via v_wmma_f32_16x16x32_bf16.
// c lives in VGPRs for the whole scan; h ping-pongs through LDS.
// A fragment = two contiguous 16B runs -> 2x ds_load_b128 per fragment.
// h writeback to Y is a cooperative coalesced LDS->global b128 copy.
// ---------------------------------------------------------------------------
__global__ __launch_bounds__(256) void enc_scan_kernel(
    const unsigned short* __restrict__ Xin, int Din,
    const unsigned short* __restrict__ Wpack, size_t wPerDir,
    const float* __restrict__ bias2,        // [2][1024]
    unsigned short* __restrict__ Y,         // [T][B][512] bf16
    unsigned short* __restrict__ Hfin,      // [B][512] bf16 (decoder h0)
    float* __restrict__ Cfin) {             // [B][512] f32  (decoder c0)
  __shared__ __align__(16) unsigned short xstage[BT * 512];
  __shared__ __align__(16) unsigned short hlds[BT * 256];
  const int dir = blockIdx.x & 1;
  const int b0  = (blockIdx.x >> 1) * BT;
  const unsigned short* W = Wpack + (size_t)dir * wPerDir;
  const float* bs = bias2 + dir * 1024;
  const int tid = threadIdx.x, w = tid >> 5, lane = tid & 31;
  const int hi = lane >> 4, ln = lane & 15;
  const int KT = (Din + HH) >> 5;

  for (int i = tid; i < BT * 256 / 8; i += 256) ((v4u*)hlds)[i] = (v4u)0u;

  v8f acc[2][4];
  float creg[2][8], hreg[2][8];
#pragma unroll
  for (int a = 0; a < 2; ++a)
#pragma unroll
    for (int r = 0; r < 8; ++r) { creg[a][r] = 0.f; hreg[a][r] = 0.f; }
  __syncthreads();

  for (int tt = 0; tt < TT; ++tt) {
    const int t = dir ? (TT - 1 - tt) : tt;
    {  // stage x_t tile (bf16) into LDS, 16B chunks
      const v4u* src = (const v4u*)(Xin + (size_t)(t * BN + b0) * Din);
      v4u* dst = (v4u*)xstage;
      const int n128 = (BT * Din) >> 3;
      for (int i = tid; i < n128; i += 256) dst[i] = src[i];
    }
    __syncthreads();

#pragma unroll
    for (int jtl = 0; jtl < 2; ++jtl) {
      const int j = ((2 * w + jtl) << 4) + ln;
#pragma unroll
      for (int g = 0; g < 4; ++g) {
        const float bv = bs[g * 256 + j];
#pragma unroll
        for (int r = 0; r < 8; ++r) acc[jtl][g][r] = bv;
      }
    }

    for (int kt = 0; kt < KT; ++kt) {
      const int kb = kt * 32 + 8 * hi;
      // k-tile never straddles the x/h boundary (Din % 32 == 0)
      const unsigned short* base = (kt * 32 < Din)
          ? &xstage[ln * Din + kb] : &hlds[(ln << 8) + (kb - Din)];
      AFrag af;
      af.q[0] = *(const v4u*)base;          // k = kb .. kb+7
      af.q[1] = *(const v4u*)(base + 16);   // k = kb+16 .. kb+23
      const v16bf a = __builtin_bit_cast(v16bf, af.o);
#pragma unroll
      for (int jtl = 0; jtl < 2; ++jtl) {
#pragma unroll
        for (int g = 0; g < 4; ++g) {
          const int nt = (g << 4) + 2 * w + jtl;
          const v8u* wp = (const v8u*)(W + (((size_t)kt * 64 + nt) << 9));
          const v16bf bfrag = __builtin_bit_cast(v16bf, wp[lane]);
          acc[jtl][g] = __builtin_amdgcn_wmma_f32_16x16x32_bf16(
              false, a, false, bfrag, (short)0, acc[jtl][g], false, false);
        }
      }
    }
    __syncthreads();  // all waves done reading hlds before overwrite

#pragma unroll
    for (int jtl = 0; jtl < 2; ++jtl) {
      const int j = ((2 * w + jtl) << 4) + ln;
#pragma unroll
      for (int r = 0; r < 8; ++r) {
        const float ig = sigm(acc[jtl][0][r]);
        const float fg = sigm(acc[jtl][1][r]);
        const float gg = tanh_(acc[jtl][2][r]);
        const float og = sigm(acc[jtl][3][r]);
        const float c = fg * creg[jtl][r] + ig * gg;
        const float h = og * tanh_(c);
        creg[jtl][r] = c;
        hreg[jtl][r] = h;
        const int m = r + (hi << 3);
        hlds[(m << 8) + j] = f2bf(h);
      }
    }
    __syncthreads();

    {  // coalesced h_t writeback: LDS -> Y[t][b0..b0+15][dir*256 ..]
      const v4u* src = (const v4u*)hlds;
      for (int i = tid; i < BT * 256 / 8; i += 256) {
        const int m = i >> 5;            // 32 x 16B per row
        const int col = (i & 31) << 3;
        *(v4u*)&Y[((size_t)t * BN + b0 + m) * (2 * HH) + (dir << 8) + col] = src[i];
      }
    }
  }

#pragma unroll
  for (int jtl = 0; jtl < 2; ++jtl) {
    const int j = ((2 * w + jtl) << 4) + ln;
#pragma unroll
    for (int r = 0; r < 8; ++r) {
      const int m = r + (hi << 3);
      Hfin[(size_t)(b0 + m) * 512 + (dir << 8) + j] = f2bf(hreg[jtl][r]);
      Cfin[(size_t)(b0 + m) * 512 + (dir << 8) + j] = creg[jtl][r];
    }
  }
}

// ---------------------------------------------------------------------------
// Decoder LSTM cell (H2=512, 2048 gate cols): one launch per (step, layer).
// gates = [z ; h] @ Wcat + bias; wave w owns N-tiles {g*32 + 4w + jtl}.
// ---------------------------------------------------------------------------
__global__ __launch_bounds__(256) void dec_cell_kernel(
    const unsigned short* __restrict__ Z, int Din,
    const unsigned short* __restrict__ Wpack,
    const float* __restrict__ bias,         // [2048]
    unsigned short* __restrict__ Hst,       // [B][512] bf16 (in/out)
    float* __restrict__ Cst) {              // [B][512] f32  (in/out)
  __shared__ __align__(16) unsigned short astage[BT * 1024];
  const int b0 = blockIdx.x * BT;
  const int tid = threadIdx.x, w = tid >> 5, lane = tid & 31;
  const int hi = lane >> 4, ln = lane & 15;
  const int K = Din + 512;
  const int KT = K >> 5;

  {  // stage [z ; h] rows, 16B chunks (Din % 8 == 0)
    const int KQ = K >> 3;
    for (int i = tid; i < BT * KQ; i += 256) {
      const int m = i / KQ, k = (i - m * KQ) << 3;
      v4u val;
      if (k < Din) val = *(const v4u*)&Z[(size_t)(b0 + m) * Din + k];
      else         val = *(const v4u*)&Hst[(size_t)(b0 + m) * 512 + (k - Din)];
      *(v4u*)&astage[m * K + k] = val;
    }
  }
  __syncthreads();

  v8f acc[4][4];
  float creg[4][8];
#pragma unroll
  for (int jtl = 0; jtl < 4; ++jtl) {
    const int j = ((4 * w + jtl) << 4) + ln;
#pragma unroll
    for (int g = 0; g < 4; ++g) {
      const float bv = bias[g * 512 + j];
#pragma unroll
      for (int r = 0; r < 8; ++r) acc[jtl][g][r] = bv;
    }
#pragma unroll
    for (int r = 0; r < 8; ++r) {
      const int m = r + (hi << 3);
      creg[jtl][r] = Cst[(size_t)(b0 + m) * 512 + j];
    }
  }

  for (int kt = 0; kt < KT; ++kt) {
    const int kb = kt * 32 + 8 * hi;
    const unsigned short* base = &astage[ln * K + kb];
    AFrag af;
    af.q[0] = *(const v4u*)base;
    af.q[1] = *(const v4u*)(base + 16);
    const v16bf a = __builtin_bit_cast(v16bf, af.o);
#pragma unroll
    for (int jtl = 0; jtl < 4; ++jtl) {
#pragma unroll
      for (int g = 0; g < 4; ++g) {
        const int nt = (g << 5) + 4 * w + jtl;
        const v8u* wp = (const v8u*)(Wpack + (((size_t)kt * 128 + nt) << 9));
        const v16bf bfrag = __builtin_bit_cast(v16bf, wp[lane]);
        acc[jtl][g] = __builtin_amdgcn_wmma_f32_16x16x32_bf16(
            false, a, false, bfrag, (short)0, acc[jtl][g], false, false);
      }
    }
  }

#pragma unroll
  for (int jtl = 0; jtl < 4; ++jtl) {
    const int j = ((4 * w + jtl) << 4) + ln;
#pragma unroll
    for (int r = 0; r < 8; ++r) {
      const float ig = sigm(acc[jtl][0][r]);
      const float fg = sigm(acc[jtl][1][r]);
      const float gg = tanh_(acc[jtl][2][r]);
      const float og = sigm(acc[jtl][3][r]);
      const float c = fg * creg[jtl][r] + ig * gg;
      const float h = og * tanh_(c);
      const int m = r + (hi << 3);
      Cst[(size_t)(b0 + m) * 512 + j] = c;
      Hst[(size_t)(b0 + m) * 512 + j] = f2bf(h);
    }
  }
}

// ---------------------------------------------------------------------------
// FC: out = z @ fc_W.T + fc_b (256x64x512) via WMMA; writes f32 d_out slice
// and bf16 feedback Znext for the next decoder step.
// ---------------------------------------------------------------------------
__global__ __launch_bounds__(128) void fc_kernel(
    const unsigned short* __restrict__ Z,    // [B][512] bf16
    const unsigned short* __restrict__ Wp,   // packed KT=16, NT=4
    const float* __restrict__ fcb,           // [64]
    float* __restrict__ out,                 // [B][P][64]
    unsigned short* __restrict__ Znext,      // [B][64]
    int p) {
  const int b0 = blockIdx.x * BT;
  const int tid = threadIdx.x, w = tid >> 5, lane = tid & 31;
  const int hi = lane >> 4, ln = lane & 15;
  v8f acc;
  const float bv = fcb[(w << 4) + ln];
#pragma unroll
  for (int r = 0; r < 8; ++r) acc[r] = bv;

  for (int kt = 0; kt < 16; ++kt) {
    const int kb = kt * 32 + 8 * hi;
    const unsigned short* base = &Z[(size_t)(b0 + ln) * 512 + kb];
    AFrag af;
    af.q[0] = *(const v4u*)base;
    af.q[1] = *(const v4u*)(base + 16);
    const v16bf a = __builtin_bit_cast(v16bf, af.o);
    const v8u* wp = (const v8u*)(Wp + (((size_t)kt * 4 + w) << 9));
    const v16bf bfrag = __builtin_bit_cast(v16bf, wp[lane]);
    acc = __builtin_amdgcn_wmma_f32_16x16x32_bf16(false, a, false, bfrag,
                                                  (short)0, acc, false, false);
  }
#pragma unroll
  for (int r = 0; r < 8; ++r) {
    const int m = r + (hi << 3);
    const int n = (w << 4) + ln;
    const float h = acc[r];
    out[((size_t)(b0 + m) * PP + p) * 64 + n] = h;
    Znext[(b0 + m) * 64 + n] = f2bf(h);
  }
}

// ---------------------------------------------------------------------------
extern "C" void kernel_launch(void* const* d_in, const int* in_sizes, int n_in,
                              void* d_out, int out_size, void* d_ws, size_t ws_size,
                              hipStream_t stream) {
  (void)in_sizes; (void)n_in; (void)out_size; (void)ws_size;
  const float* x         = (const float*)d_in[0];
  const float* enc_Wih0  = (const float*)d_in[1];
  const float* enc_Whh0  = (const float*)d_in[2];
  const float* enc_bih0  = (const float*)d_in[3];
  const float* enc_bhh0  = (const float*)d_in[4];
  const float* enc_Wih   = (const float*)d_in[5];
  const float* enc_Whh   = (const float*)d_in[6];
  const float* enc_bih   = (const float*)d_in[7];
  const float* enc_bhh   = (const float*)d_in[8];
  const float* dec_Wih0  = (const float*)d_in[9];
  const float* dec_Whh0  = (const float*)d_in[10];
  const float* dec_bih0  = (const float*)d_in[11];
  const float* dec_bhh0  = (const float*)d_in[12];
  const float* dec_Wih   = (const float*)d_in[13];
  const float* dec_Whh   = (const float*)d_in[14];
  const float* dec_bih   = (const float*)d_in[15];
  const float* dec_bhh   = (const float*)d_in[16];
  const float* fc_W      = (const float*)d_in[17];
  const float* fc_b      = (const float*)d_in[18];
  float* out = (float*)d_out;

  char* p = (char*)d_ws;
  auto carve = [&](size_t bytes) -> char* {
    char* r = p; p += (bytes + 255) & ~(size_t)255; return r;
  };

  const int KT0  = (DD + HH) / 32;        // 10
  const int KTl  = (2 * HH + HH) / 32;    // 24
  const size_t encW0perDir = (size_t)KT0 * 64 * 512;
  const size_t encWlperDir = (size_t)KTl * 64 * 512;
  unsigned short* encWp[LL];
  encWp[0] = (unsigned short*)carve(2 * encW0perDir * 2);
  for (int l = 1; l < LL; ++l) encWp[l] = (unsigned short*)carve(2 * encWlperDir * 2);
  float* encB = (float*)carve((size_t)LL * 2 * 1024 * 4);

  const int dKT0 = (DD + 512) / 32;       // 18
  const int dKTl = (512 + 512) / 32;      // 32
  const size_t decW0 = (size_t)dKT0 * 128 * 512;
  const size_t decWl = (size_t)dKTl * 128 * 512;
  unsigned short* decWp[LL];
  decWp[0] = (unsigned short*)carve(decW0 * 2);
  for (int l = 1; l < LL; ++l) decWp[l] = (unsigned short*)carve(decWl * 2);
  float* decB = (float*)carve((size_t)LL * 2048 * 4);
  unsigned short* fcWp = (unsigned short*)carve((size_t)16 * 4 * 512 * 2);

  unsigned short* Xa   = (unsigned short*)carve((size_t)TT * BN * 512 * 2);
  unsigned short* Xb   = (unsigned short*)carve((size_t)TT * BN * 512 * 2);
  unsigned short* Hst  = (unsigned short*)carve((size_t)LL * BN * 512 * 2);
  float*          Cst  = (float*)carve((size_t)LL * BN * 512 * 4);
  unsigned short* Zbuf = (unsigned short*)carve((size_t)BN * DD * 2);

  auto packLaunch = [&](const float* Wih, const float* Whh, int Din, int H,
                        int NT, int KT, unsigned short* dst) {
    int total = KT * NT * 512;
    pack_wcat_kernel<<<(total + 255) / 256, 256, 0, stream>>>(Wih, Whh, Din, H, NT, KT, dst);
  };

  // --- weight packing (per launch; deterministic) ---
  for (int d = 0; d < 2; ++d)
    packLaunch(enc_Wih0 + (size_t)d * 1024 * DD, enc_Whh0 + (size_t)d * 1024 * HH,
               DD, HH, 64, KT0, encWp[0] + (size_t)d * encW0perDir);
  for (int l = 1; l < LL; ++l)
    for (int d = 0; d < 2; ++d) {
      size_t wi = (size_t)(l - 1) * 2 + d;
      packLaunch(enc_Wih + wi * 1024 * (2 * HH), enc_Whh + wi * 1024 * HH,
                 2 * HH, HH, 64, KTl, encWp[l] + (size_t)d * encWlperDir);
    }
  packLaunch(dec_Wih0, dec_Whh0, DD, 512, 128, dKT0, decWp[0]);
  for (int l = 1; l < LL; ++l)
    packLaunch(dec_Wih + (size_t)(l - 1) * 2048 * 512,
               dec_Whh + (size_t)(l - 1) * 2048 * 512, 512, 512, 128, dKTl, decWp[l]);
  packLaunch(nullptr, fc_W, 0, 512, 4, 16, fcWp);

  bias_add_kernel<<<(2048 + 255) / 256, 256, 0, stream>>>(enc_bih0, enc_bhh0, encB, 2048);
  bias_add_kernel<<<(8192 + 255) / 256, 256, 0, stream>>>(enc_bih, enc_bhh, encB + 2048, 8192);
  bias_add_kernel<<<(2048 + 255) / 256, 256, 0, stream>>>(dec_bih0, dec_bhh0, decB, 2048);
  bias_add_kernel<<<(8192 + 255) / 256, 256, 0, stream>>>(dec_bih, dec_bhh, decB + 2048, 8192);

  x_prep_kernel<<<(BN * TT * DD + 255) / 256, 256, 0, stream>>>(x, Xa, Zbuf);

  // --- encoder: 5 persistent-scan launches (fwd+bwd blocks concurrent) ---
  unsigned short* bufs[2] = {Xa, Xb};
  for (int l = 0; l < LL; ++l) {
    const int Din = l ? 2 * HH : DD;
    const size_t perDir = l ? encWlperDir : encW0perDir;
    enc_scan_kernel<<<(BN / BT) * 2, 256, 0, stream>>>(
        bufs[l & 1], Din, encWp[l], perDir, encB + (size_t)l * 2048,
        bufs[(l + 1) & 1], Hst + (size_t)l * BN * 512, Cst + (size_t)l * BN * 512);
  }

  // --- decoder: per-(step,layer) WMMA cells + WMMA FC with feedback ---
  for (int ps = 0; ps < PP; ++ps) {
    for (int l = 0; l < LL; ++l) {
      const unsigned short* Zin = l ? (Hst + (size_t)(l - 1) * BN * 512) : Zbuf;
      const int Din = l ? 512 : DD;
      dec_cell_kernel<<<BN / BT, 256, 0, stream>>>(
          Zin, Din, decWp[l], decB + (size_t)l * 2048,
          Hst + (size_t)l * BN * 512, Cst + (size_t)l * BN * 512);
    }
    fc_kernel<<<BN / BT, 128, 0, stream>>>(Hst + (size_t)4 * BN * 512, fcWp, fc_b,
                                           out, Zbuf, ps);
  }
}
